// DifferentialAttention_59596966199767
// MI455X (gfx1250) — matrix operand order repaired and checked by reading, not compiled, not run on hardware
//
#include <hip/hip_runtime.h>
#include <hip/hip_bf16.h>

#define S 2048
#define D 2048
#define H 16
#define KV 4
#define HD 128
#define HALF 64
#define KDIM 2048   // contraction depth of every WMMA GEMM in this pipeline

typedef __attribute__((ext_vector_type(16))) __bf16 bf16x16;
typedef __attribute__((ext_vector_type(8)))  __bf16 bf16x8;
typedef __attribute__((ext_vector_type(8)))  float  f32x8;

// A-fragment / LDS-P fragment: elements [0..7] at p, [8..15] at p+16
static __device__ __forceinline__ bf16x16 ld_split16(const __bf16* p) {
  bf16x8 lo = *(const bf16x8*)p;
  bf16x8 hi = *(const bf16x8*)(p + 16);
  bf16x16 r;
#pragma unroll
  for (int i = 0; i < 8; ++i) { r[i] = lo[i]; r[i + 8] = hi[i]; }
  return r;
}
// B-fragment: 16 contiguous bf16
static __device__ __forceinline__ bf16x16 ld_contig16(const __bf16* p) {
  bf16x8 lo = *(const bf16x8*)p;
  bf16x8 hi = *(const bf16x8*)(p + 8);
  bf16x16 r;
#pragma unroll
  for (int i = 0; i < 8; ++i) { r[i] = lo[i]; r[i + 8] = hi[i]; }
  return r;
}

static __device__ __forceinline__ f32x8 wmma_bf16(bf16x16 a, bf16x16 b, f32x8 c) {
  return __builtin_amdgcn_wmma_f32_16x16x32_bf16(false, a, false, b, (short)0, c,
                                                 false, false);
}

// CDNA5 async global->LDS copy (ASYNCcnt-tracked DMA, 16B per lane).
static __device__ __forceinline__ void async_copy_b128(uint32_t lds_off,
                                                       const void* gaddr) {
  asm volatile("global_load_async_to_lds_b128 %0, %1, off"
               :: "v"(lds_off), "v"(gaddr) : "memory");
}
static __device__ __forceinline__ void wait_async_le6() {
  asm volatile("s_wait_asynccnt 0x6" ::: "memory");
}
static __device__ __forceinline__ void wait_async_0() {
  asm volatile("s_wait_asynccnt 0x0" ::: "memory");
}

static __device__ __forceinline__ float redmax16(float v) {
#pragma unroll
  for (int m = 1; m <= 8; m <<= 1) v = fmaxf(v, __shfl_xor(v, m, 32));
  return v;
}
static __device__ __forceinline__ float redsum16(float v) {
#pragma unroll
  for (int m = 1; m <= 8; m <<= 1) v += __shfl_xor(v, m, 32);
  return v;
}

// ---------------------------------------------------------------- elementwise
__global__ void convert_f32_bf16(const float* __restrict__ src,
                                 __bf16* __restrict__ dst, int n) {
  int i = blockIdx.x * blockDim.x + threadIdx.x;
  if (i < n) dst[i] = (__bf16)src[i];
}

// src fp32 [Kd][Nd] -> dst bf16 [Nd][Kd], LDS-tiled 32x32
__global__ void transpose_convert(const float* __restrict__ src,
                                  __bf16* __restrict__ dst, int Kd, int Nd) {
  __shared__ float tile[32][33];
  int nb = Nd >> 5;
  int bx = blockIdx.x % nb;
  int by = blockIdx.x / nb;
  int tx = threadIdx.x & 31;
  int ty = threadIdx.x >> 5; // 0..7
#pragma unroll
  for (int r = ty; r < 32; r += 8)
    tile[r][tx] = src[(size_t)(by * 32 + r) * Nd + bx * 32 + tx];
  __syncthreads();
#pragma unroll
  for (int r = ty; r < 32; r += 8)
    dst[(size_t)(bx * 32 + r) * Kd + by * 32 + tx] = (__bf16)tile[tx][r];
}

// Vf fp32 [S][KV*HD] -> Vt bf16 [KV][HD][S]
__global__ void v_transpose(const float* __restrict__ Vf, __bf16* __restrict__ Vt) {
  int i = blockIdx.x * blockDim.x + threadIdx.x;
  if (i >= S * KV * HD) return;
  int d = i % HD;
  int kv = (i / HD) % KV;
  int s = i / (HD * KV);
  Vt[((size_t)kv * HD + d) * S + s] = (__bf16)Vf[i];
}

// src fp32 [S][nh][HD] -> rope -> dst bf16 [nh][S][HD]
__global__ void rope_convert(const float* __restrict__ src,
                             const float* __restrict__ cosb,
                             const float* __restrict__ sinb,
                             __bf16* __restrict__ dst, int nh) {
  int i = blockIdx.x * blockDim.x + threadIdx.x;
  if (i >= S * nh * HALF) return;
  int p = i % HALF;
  int h = (i / HALF) % nh;
  int s = i / (HALF * nh);
  float c = cosb[s * HALF + p];
  float sn = sinb[s * HALF + p];
  float x0 = src[((size_t)s * nh + h) * HD + 2 * p];
  float x1 = src[((size_t)s * nh + h) * HD + 2 * p + 1];
  dst[((size_t)h * S + s) * HD + 2 * p]     = (__bf16)(x0 * c - x1 * sn);
  dst[((size_t)h * S + s) * HD + 2 * p + 1] = (__bf16)(x0 * sn + x1 * c);
}

// ---------------------------------------------------------------- WMMA GEMM
// C[M][N] fp32 = A[M][KDIM] bf16 (row major) * Bt[N][KDIM] bf16 (B transposed).
// Block = 128x64 macro tile, 8 waves (each 16x64). Both A (128x64) and B
// (64x64) k-panels are staged in LDS by the async global->LDS DMA engine,
// double buffered (6 async ops per thread per stage), so the steady-state
// loop is branch-free: issue next stage, s_wait_asynccnt<=6, barrier,
// 8 WMMAs from LDS, barrier.
__global__ void __launch_bounds__(256)
gemm_bf16_wmma(const __bf16* __restrict__ A, const __bf16* __restrict__ Bt,
               float* __restrict__ C, int M, int N) {
  __shared__ alignas(16) __bf16 astage[2][128 * 64];
  __shared__ alignas(16) __bf16 bstage[2][64 * 64];
  const int tid = threadIdx.x;
  const int wid = tid >> 5, lane = tid & 31;
  const int grp = lane >> 4, ln = lane & 15;
  const int nblk = N >> 6;
  const int mb = blockIdx.x / nblk;
  const int nb = blockIdx.x % nblk;

  // A staging: thread copies 64B (row = tid>>1, 32 elems at (tid&1)*32)
  const int ar = tid >> 1, ac = (tid & 1) * 32;
  const __bf16* asrc = A + (size_t)(mb * 128 + ar) * KDIM + ac;
  // B staging: thread copies 32B (row = tid>>2, 16 elems at (tid&3)*16)
  const int br = tid >> 2, bc = (tid & 3) * 16;
  const __bf16* bsrc = Bt + (size_t)(nb * 64 + br) * KDIM + bc;
  uint32_t adst[2], bdst[2];
#pragma unroll
  for (int b = 0; b < 2; ++b) {
    adst[b] = (uint32_t)(size_t)&astage[b][ar * 64 + ac];
    bdst[b] = (uint32_t)(size_t)&bstage[b][br * 64 + bc];
  }

  auto stage = [&](int step) {
    int buf = step & 1;
    const __bf16* as = asrc + step * 64;
    const __bf16* bs = bsrc + step * 64;
#pragma unroll
    for (int c = 0; c < 4; ++c)
      async_copy_b128(adst[buf] + c * 16, as + c * 8);
#pragma unroll
    for (int c = 0; c < 2; ++c)
      async_copy_b128(bdst[buf] + c * 16, bs + c * 8);
  };

  f32x8 acc[4] = {};
  auto compute = [&](const __bf16* abuf, const __bf16* bbuf) {
    bf16x16 a0 = ld_contig16(abuf + (wid * 16 + ln) * 64 + grp * 16);
    bf16x16 a1 = ld_contig16(abuf + (wid * 16 + ln) * 64 + 32 + grp * 16);
#pragma unroll
    for (int j = 0; j < 4; ++j) {
      bf16x16 b0 = ld_contig16(bbuf + (j * 16 + ln) * 64 + grp * 16);
      bf16x16 b1 = ld_contig16(bbuf + (j * 16 + ln) * 64 + 32 + grp * 16);
      acc[j] = wmma_bf16(a0, b0, acc[j]);
      acc[j] = wmma_bf16(a1, b1, acc[j]);
    }
  };

  const int NK = KDIM / 64; // 32 k-steps of depth 64
  stage(0);
#pragma unroll 1
  for (int i = 0; i < NK - 1; ++i) {
    stage(i + 1);        // fill the other buffer
    wait_async_le6();    // only the OLDER stage (6 ops) must have landed
    __syncthreads();
    compute(astage[i & 1], bstage[i & 1]);
    __syncthreads();     // all reads done before this buffer is re-filled
  }
  wait_async_0();        // epilogue peeled: final k-step, no new stage
  __syncthreads();
  compute(astage[(NK - 1) & 1], bstage[(NK - 1) & 1]);

  const int row0 = mb * 128 + wid * 16;
#pragma unroll
  for (int j = 0; j < 4; ++j)
#pragma unroll
    for (int r = 0; r < 8; ++r)
      C[(size_t)(row0 + r + grp * 8) * N + nb * 64 + j * 16 + ln] = acc[j][r];
}

// ---------------------------------------------------------------- attention
// One wave per (head, 16-row q tile); flash over 32-key steps; two softmax
// streams (differential halves) sharing the V fragments.
__global__ void __launch_bounds__(128)
diff_attn_kernel(const __bf16* __restrict__ Qb, const __bf16* __restrict__ Kb,
                 const __bf16* __restrict__ Vt, const float* __restrict__ lambda_p,
                 __bf16* __restrict__ attn_out) {
  __shared__ alignas(16) __bf16 plds[4][2][16 * 40];
  int wid = threadIdx.x >> 5, lane = threadIdx.x & 31;
  int grp = lane >> 4, ln = lane & 15;
  int task = blockIdx.x * 4 + wid;
  int qt = task & 127;
  int h = task >> 7;
  int qs = qt * 16;
  int kvh = h >> 2;
  const __bf16* Qh = Qb + (size_t)h * S * HD;
  const __bf16* Kh = Kb + (size_t)kvh * S * HD;
  const __bf16* Vh = Vt + (size_t)kvh * HD * S;
  __bf16* pl1 = plds[wid][0];
  __bf16* pl2 = plds[wid][1];

  bf16x16 q1[2], q2[2];
#pragma unroll
  for (int i = 0; i < 2; ++i) {
    q1[i] = ld_contig16(Qh + (size_t)(qs + ln) * HD + i * 32 + grp * 16);
    q2[i] = ld_contig16(Qh + (size_t)(qs + ln) * HD + 64 + i * 32 + grp * 16);
  }

  f32x8 acc1[8] = {}, acc2[8] = {};
  float m1[8], l1[8], m2[8], l2[8];
#pragma unroll
  for (int r = 0; r < 8; ++r) {
    m1[r] = -__builtin_inff(); l1[r] = 0.f;
    m2[r] = -__builtin_inff(); l2[r] = 0.f;
  }
  const float scale = 0.125f; // 1/sqrt(HALF)

  for (int ks = 0; ks < qs + 16; ks += 32) {
    f32x8 s1[2] = {}, s2[2] = {};
#pragma unroll
    for (int n = 0; n < 2; ++n) {
      const __bf16* kp = Kh + (size_t)(ks + n * 16 + ln) * HD + grp * 16;
#pragma unroll
      for (int i = 0; i < 2; ++i) {
        s1[n] = wmma_bf16(q1[i], ld_contig16(kp + i * 32), s1[n]);
        s2[n] = wmma_bf16(q2[i], ld_contig16(kp + 64 + i * 32), s2[n]);
      }
    }
    // causal mask + scale
#pragma unroll
    for (int n = 0; n < 2; ++n)
#pragma unroll
      for (int r = 0; r < 8; ++r) {
        int row = qs + r + grp * 8;
        int key = ks + n * 16 + ln;
        bool ok = key <= row;
        s1[n][r] = ok ? s1[n][r] * scale : -__builtin_inff();
        s2[n][r] = ok ? s2[n][r] * scale : -__builtin_inff();
      }
    // online softmax per row (16-lane reductions), P -> LDS as bf16
#pragma unroll
    for (int r = 0; r < 8; ++r) {
      int row = r + grp * 8;
      {
        float mx = redmax16(fmaxf(s1[0][r], s1[1][r]));
        float mn = fmaxf(m1[r], mx);
        float corr = __expf(m1[r] - mn);
        float p0 = __expf(s1[0][r] - mn);
        float p1 = __expf(s1[1][r] - mn);
        l1[r] = l1[r] * corr + redsum16(p0 + p1);
        m1[r] = mn;
#pragma unroll
        for (int t = 0; t < 8; ++t) acc1[t][r] *= corr;
        pl1[row * 40 + ln]      = (__bf16)p0;
        pl1[row * 40 + 16 + ln] = (__bf16)p1;
      }
      {
        float mx = redmax16(fmaxf(s2[0][r], s2[1][r]));
        float mn = fmaxf(m2[r], mx);
        float corr = __expf(m2[r] - mn);
        float p0 = __expf(s2[0][r] - mn);
        float p1 = __expf(s2[1][r] - mn);
        l2[r] = l2[r] * corr + redsum16(p0 + p1);
        m2[r] = mn;
#pragma unroll
        for (int t = 0; t < 8; ++t) acc2[t][r] *= corr;
        pl2[row * 40 + ln]      = (__bf16)p0;
        pl2[row * 40 + 16 + ln] = (__bf16)p1;
      }
    }
    // P (C-layout) -> A-layout via per-wave LDS (intra-wave DS ordering)
    bf16x16 pf1 = ld_contig16(pl1 + ln * 40 + grp * 16);
    bf16x16 pf2 = ld_contig16(pl2 + ln * 40 + grp * 16);
#pragma unroll
    for (int t = 0; t < 8; ++t) {
      const __bf16* vp = Vh + (size_t)(t * 16 + ln) * S + ks + grp * 16;
      bf16x16 bv = ld_contig16(vp);
      acc1[t] = wmma_bf16(pf1, bv, acc1[t]);
      acc2[t] = wmma_bf16(pf2, bv, acc2[t]);
    }
  }

  float lam = 1.f / (1.f + __expf(-lambda_p[h]));
#pragma unroll
  for (int t = 0; t < 8; ++t)
#pragma unroll
    for (int r = 0; r < 8; ++r) {
      float o = (acc1[t][r] / l1[r] - lam * acc2[t][r] / l2[r]) * 0.5f;
      attn_out[(size_t)(qs + r + grp * 8) * (H * HD) + h * HD + t * 16 + ln] =
          (__bf16)o;
    }
}

// ---------------------------------------------------------------- launch
extern "C" void kernel_launch(void* const* d_in, const int* in_sizes, int n_in,
                              void* d_out, int out_size, void* d_ws, size_t ws_size,
                              hipStream_t stream) {
  (void)in_sizes; (void)n_in; (void)out_size; (void)ws_size;
  const float* x    = (const float*)d_in[0];
  const float* fcos = (const float*)d_in[1];
  const float* fsin = (const float*)d_in[2];
  const float* Wq   = (const float*)d_in[3];
  const float* Wk   = (const float*)d_in[4];
  const float* Wv   = (const float*)d_in[5];
  const float* Wo   = (const float*)d_in[6];
  const float* lam  = (const float*)d_in[7];

  char* ws = (char*)d_ws;
  size_t off = 0;
  auto alloc = [&](size_t bytes) {
    void* p = ws + off;
    off += (bytes + 255) & ~(size_t)255;
    return p;
  };
  __bf16* xb  = (__bf16*)alloc((size_t)S * D * 2);
  __bf16* Wqt = (__bf16*)alloc((size_t)D * H * HD * 2);
  __bf16* Wkt = (__bf16*)alloc((size_t)D * KV * HD * 2);
  __bf16* Wvt = (__bf16*)alloc((size_t)D * KV * HD * 2);
  __bf16* Wot = (__bf16*)alloc((size_t)H * HD * D * 2);
  float*  Qf  = (float*)alloc((size_t)S * H * HD * 4);
  float*  Kf  = (float*)alloc((size_t)S * KV * HD * 4);
  float*  Vf  = (float*)alloc((size_t)S * KV * HD * 4);
  __bf16* Qbr = (__bf16*)alloc((size_t)H * S * HD * 2);
  __bf16* Kbr = (__bf16*)alloc((size_t)KV * S * HD * 2);
  __bf16* Vtb = (__bf16*)alloc((size_t)KV * HD * S * 2);
  __bf16* Att = (__bf16*)alloc((size_t)S * H * HD * 2);

  convert_f32_bf16<<<(S * D) / 256, 256, 0, stream>>>(x, xb, S * D);
  transpose_convert<<<(D / 32) * ((H * HD) / 32), 256, 0, stream>>>(Wq, Wqt, D, H * HD);
  transpose_convert<<<(D / 32) * ((KV * HD) / 32), 256, 0, stream>>>(Wk, Wkt, D, KV * HD);
  transpose_convert<<<(D / 32) * ((KV * HD) / 32), 256, 0, stream>>>(Wv, Wvt, D, KV * HD);
  transpose_convert<<<((H * HD) / 32) * (D / 32), 256, 0, stream>>>(Wo, Wot, H * HD, D);

  gemm_bf16_wmma<<<(S / 128) * ((H * HD) / 64), 256, 0, stream>>>(xb, Wqt, Qf, S, H * HD);
  gemm_bf16_wmma<<<(S / 128) * ((KV * HD) / 64), 256, 0, stream>>>(xb, Wkt, Kf, S, KV * HD);
  gemm_bf16_wmma<<<(S / 128) * ((KV * HD) / 64), 256, 0, stream>>>(xb, Wvt, Vf, S, KV * HD);

  rope_convert<<<(S * H * HALF) / 256, 256, 0, stream>>>(Qf, fcos, fsin, Qbr, H);
  rope_convert<<<(S * KV * HALF) / 256, 256, 0, stream>>>(Kf, fcos, fsin, Kbr, KV);
  v_transpose<<<(S * KV * HD) / 256, 256, 0, stream>>>(Vf, Vtb);

  diff_attn_kernel<<<(H * (S / 16)) / 4, 128, 0, stream>>>(Qbr, Kbr, Vtb, lam, Att);

  gemm_bf16_wmma<<<(S / 128) * (D / 64), 256, 0, stream>>>(Att, Wot, (float*)d_out, S, D);
}
